// GRU_28913719837003
// MI455X (gfx1250) — matrix operand order repaired and checked by reading, not compiled, not run on hardware
//
#include <hip/hip_runtime.h>

// GRU cell, fully fused: two bf16 WMMA GEMMs (x@Wi^T, h@Wh^T) + gate math.
// B=8192, IN=H=2048. Block = 256 threads (8 wave32) -> 64x64 tile of ht.

#define KD 2048           // reduction dim (IN == H == 2048)
#define HD 2048           // output cols
#define TM 64             // block tile rows (batch)
#define TN 64             // block tile cols (hidden units)
#define KS 32             // K per WMMA step
#define LDS_STRIDE 40     // 32 + 8 pad, in bf16 elems
#define TILE_ELEMS (64 * LDS_STRIDE)

typedef __attribute__((ext_vector_type(16))) __bf16 v16bf;
typedef __attribute__((ext_vector_type(8)))  __bf16 v8bf;
typedef __attribute__((ext_vector_type(4)))  __bf16 v4bf;
typedef __attribute__((ext_vector_type(8)))  float  v8f;

__device__ __forceinline__ v16bf ld_frag(const unsigned short* p0,
                                         const unsigned short* p1) {
  v8bf lo = *(const v8bf*)p0;        // ds_read_b128
  v8bf hi = *(const v8bf*)p1;        // ds_read_b128
  v16bf r;
#pragma unroll
  for (int i = 0; i < 8; ++i) { r[i] = lo[i]; r[i + 8] = hi[i]; }
  return r;
}

__global__ __launch_bounds__(256) void gru_fused_wmma(
    const float* __restrict__ x,   const float* __restrict__ h,
    const float* __restrict__ wir, const float* __restrict__ bir,
    const float* __restrict__ whr, const float* __restrict__ bhr,
    const float* __restrict__ wiz, const float* __restrict__ biz,
    const float* __restrict__ whz, const float* __restrict__ bhz,
    const float* __restrict__ win, const float* __restrict__ bin,
    const float* __restrict__ whn, const float* __restrict__ bhn,
    float* __restrict__ out)
{
  __shared__ unsigned short lds[8 * TILE_ELEMS];   // 40 KB

  const int tid   = threadIdx.x;
  const int lane  = tid & 31;
  const int wid   = tid >> 5;
  const int waveM = wid & 3;    // 4 waves along M -> 16 rows each
  const int waveN = wid >> 2;   // 2 waves along N -> 32 cols each
  const int half  = lane >> 4;  // half-wave select (ISA bf16 layouts)
  const int l16   = lane & 15;

  const int rowBase = blockIdx.y * TM;   // batch rows
  const int colBase = blockIdx.x * TN;   // hidden units

  // Tile sources: 0:x 1:h 2:Wir 3:Wiz 4:Win 5:Whr 6:Whz 7:Whn
  const float* srcs[8] = {
    x   + (size_t)rowBase * KD,
    h   + (size_t)rowBase * KD,
    wir + (size_t)colBase * KD,
    wiz + (size_t)colBase * KD,
    win + (size_t)colBase * KD,
    whr + (size_t)colBase * KD,
    whz + (size_t)colBase * KD,
    whn + (size_t)colBase * KD,
  };

  v8f acc[6][2];
#pragma unroll
  for (int g = 0; g < 6; ++g)
#pragma unroll
    for (int s = 0; s < 2; ++s) {
      v8f z = {};
      acc[g][s] = z;
    }

  const int kgrp = tid & 7;    // 4-float chunk within K-step
  const int rsub = tid >> 3;   // rows rsub, rsub+32

  for (int k0 = 0; k0 < KD; k0 += KS) {
    __syncthreads();
    // ---- stage 8 tiles: fp32 global -> bf16 LDS (64 rows x 32 k each) ----
#pragma unroll
    for (int t = 0; t < 8; ++t) {
      const float* src = srcs[t] + k0 + kgrp * 4;
#pragma unroll
      for (int rr = 0; rr < 2; ++rr) {
        const int r = rsub + rr * 32;
        float4 v = *(const float4*)(src + (size_t)r * KD);   // global_load_b128
        v4bf p;                      // native v_cvt f32->bf16 (RNE)
        p[0] = (__bf16)v.x;
        p[1] = (__bf16)v.y;
        p[2] = (__bf16)v.z;
        p[3] = (__bf16)v.w;
        *(v4bf*)&lds[t * TILE_ELEMS + r * LDS_STRIDE + kgrp * 4] = p; // ds_store_b64
      }
    }
    __syncthreads();

    if (k0 + KS < KD) {
#pragma unroll
      for (int t = 0; t < 8; ++t)   // global_prefetch_b8 of next K slab
        __builtin_prefetch(srcs[t] + (size_t)rsub * KD + k0 + KS + kgrp * 4, 0, 1);
    }

    // ---- A fragments (16x32 bf16, ISA half-wave K interleave) ----
    const int arow = waveM * 16 + l16;
    const unsigned short* axp = &lds[0 * TILE_ELEMS + arow * LDS_STRIDE];
    const unsigned short* ahp = &lds[1 * TILE_ELEMS + arow * LDS_STRIDE];
    const v16bf a_x = ld_frag(axp + half * 8, axp + 16 + half * 8);
    const v16bf a_h = ld_frag(ahp + half * 8, ahp + 16 + half * 8);

    // ---- 12 WMMAs: 6 gates x 2 N-subtiles ----
#pragma unroll
    for (int s = 0; s < 2; ++s) {
      const int brow = waveN * 32 + s * 16 + l16;   // column of W^T = row of W
#pragma unroll
      for (int g = 0; g < 6; ++g) {
        const unsigned short* bp =
            &lds[(2 + g) * TILE_ELEMS + brow * LDS_STRIDE + half * 8];
        const v16bf b = ld_frag(bp, bp + 16);
        const v16bf a = (g < 3) ? a_x : a_h;
        acc[g][s] = __builtin_amdgcn_wmma_f32_16x16x32_bf16(
            false, a, false, b, (short)0, acc[g][s], false, false);
      }
    }
  }

  // ---- epilogue: biases + sigmoid/tanh gate fusion, blend with hidden ----
#pragma unroll
  for (int s = 0; s < 2; ++s) {
    const int col = colBase + waveN * 32 + s * 16 + l16;
    const float vbir = bir[col], vbiz = biz[col], vbin = bin[col];
    const float vbhr = bhr[col], vbhz = bhz[col], vbhn = bhn[col];
#pragma unroll
    for (int j = 0; j < 8; ++j) {
      const int row = rowBase + waveM * 16 + half * 8 + j;  // C/D VGPR layout
      const float ir  = acc[0][s][j] + vbir;
      const float iz  = acc[1][s][j] + vbiz;
      const float inn = acc[2][s][j] + vbin;
      const float hr  = acc[3][s][j] + vbhr;
      const float hz  = acc[4][s][j] + vbhz;
      const float hn  = acc[5][s][j] + vbhn;
      const float rt = 1.0f / (1.0f + __expf(-(ir + hr)));
      const float zt = 1.0f / (1.0f + __expf(-(iz + hz)));
      const float na = inn + rt * hn;
      const float e2 = __expf(2.0f * na);
      const float nt = (e2 - 1.0f) / (e2 + 1.0f);   // tanh
      const float hv = h[(size_t)row * HD + col];
      out[(size_t)row * HD + col] = (1.0f - zt) * nt + zt * hv;
    }
  }
}

extern "C" void kernel_launch(void* const* d_in, const int* in_sizes, int n_in,
                              void* d_out, int out_size, void* d_ws, size_t ws_size,
                              hipStream_t stream) {
  (void)in_sizes; (void)n_in; (void)out_size; (void)d_ws; (void)ws_size;
  const float* x   = (const float*)d_in[0];
  const float* h   = (const float*)d_in[1];
  const float* wir = (const float*)d_in[2];
  const float* birb= (const float*)d_in[3];
  const float* whr = (const float*)d_in[4];
  const float* bhrb= (const float*)d_in[5];
  const float* wiz = (const float*)d_in[6];
  const float* bizb= (const float*)d_in[7];
  const float* whz = (const float*)d_in[8];
  const float* bhzb= (const float*)d_in[9];
  const float* win = (const float*)d_in[10];
  const float* binb= (const float*)d_in[11];
  const float* whn = (const float*)d_in[12];
  const float* bhnb= (const float*)d_in[13];
  float* out = (float*)d_out;

  dim3 grid(HD / TN, 8192 / TM);   // 32 x 128 blocks
  dim3 block(256);
  gru_fused_wmma<<<grid, block, 0, stream>>>(
      x, h, wir, birb, whr, bhrb, wiz, bizb, whz, bhzb,
      win, binb, whn, bhnb, out);
}
